// AttentionLayer_9311489098175
// MI455X (gfx1250) — matrix operand order repaired and checked by reading, not compiled, not run on hardware
//
#include <hip/hip_runtime.h>

typedef __attribute__((ext_vector_type(16))) __bf16 v16bf;
typedef __attribute__((ext_vector_type(8)))  float  v8f;

#define DEV __device__ __forceinline__

// Problem constants (from reference)
constexpr int Bc = 32;     // batch
constexpr int Tc = 8;      // new tokens
constexpr int Dm = 2048;   // d_model
constexpr int Hc = 32;     // heads
constexpr int Dh = 64;     // head dim
constexpr int CT = 1016;   // cached tokens
constexpr int Sc = 1024;   // total seq = MAX_CACHE_LEN (CT + Tc == 1024)
constexpr int Mrows = Bc * Tc;  // 256

DEV float4 ld4(const float* __restrict__ p) {
  return *reinterpret_cast<const float4*>(p);
}
DEV void st4(float* __restrict__ p, float4 v) {
  *reinterpret_cast<float4*>(p) = v;
}
DEV void put4(v16bf& v, int o, float4 f) {
  v[o]     = (__bf16)f.x;
  v[o + 1] = (__bf16)f.y;
  v[o + 2] = (__bf16)f.z;
  v[o + 3] = (__bf16)f.w;
}

// Async copy of 16 bytes from global to LDS (ASYNCcnt-tracked, in-order done).
DEV void async_ld_f4_to_lds(unsigned lds_off, const float* gptr) {
  asm volatile("global_load_async_to_lds_b128 %0, %1, off"
               :: "v"(lds_off), "v"(gptr)
               : "memory");
}
DEV void wait_async_le16() { asm volatile("s_wait_asynccnt 0x10" ::: "memory"); }
DEV void wait_async_0()    { asm volatile("s_wait_asynccnt 0x0"  ::: "memory"); }

// ---------------------------------------------------------------------------
// GEMM: out[m,n] = sum_k A[m,k] * W[n,k] + bias[n]
// A row-major [Mrows,Dm], W row-major [Dm,Dm]. One wave per 16x16 tile.
// MODE 0: out[m*Dm + n]
// MODE 1: out[((b*H+h)*T+t)*Dh + d] with m=(b,t), n=(h,d)    ([B,H,T,Dh])
// ---------------------------------------------------------------------------
template <int MODE>
__global__ __launch_bounds__(32)
void gemm_xwT_bf16(const float* __restrict__ A, const float* __restrict__ W,
                   const float* __restrict__ bias, float* __restrict__ out)
{
  const int lane  = threadIdx.x & 31;
  const int lrow  = lane & 15;
  const int khalf = lane >> 4;
  const int n0 = blockIdx.x * 16;
  const int m0 = blockIdx.y * 16;

  const float* __restrict__ arow = A + (size_t)(m0 + lrow) * Dm;
  const float* __restrict__ wrow = W + (size_t)(n0 + lrow) * Dm;

  v8f acc = {};
  for (int k0 = 0; k0 < Dm; k0 += 32) {
    __builtin_prefetch(wrow + k0 + 128, 0, 0);  // global_prefetch_b8
    const int abase = k0 + khalf * 8;   // A: lanes 0-15 K{0-7,16-23}, 16-31 +8
    const int bbase = k0 + khalf * 8;   // B: lanes 0-15 K{0-7,16-23}, 16-31 +8
    v16bf a, b;
    put4(a, 0,  ld4(arow + abase));
    put4(a, 4,  ld4(arow + abase + 4));
    put4(a, 8,  ld4(arow + abase + 16));
    put4(a, 12, ld4(arow + abase + 20));
    put4(b, 0,  ld4(wrow + bbase));
    put4(b, 4,  ld4(wrow + bbase + 4));
    put4(b, 8,  ld4(wrow + bbase + 16));
    put4(b, 12, ld4(wrow + bbase + 20));
    acc = __builtin_amdgcn_wmma_f32_16x16x32_bf16(false, a, false, b,
                                                  (short)0, acc, false, false);
  }

  const int n  = n0 + lrow;
  const float bn = bias[n];
#pragma unroll
  for (int r = 0; r < 8; ++r) {
    const int m = m0 + khalf * 8 + r;   // C/D layout: lane half selects M block
    const float v = acc[r] + bn;
    if (MODE == 0) {
      out[(size_t)m * Dm + n] = v;
    } else {
      const int bb = m >> 3, t = m & 7;          // /Tc, %Tc
      const int h  = n >> 6, d = n & 63;         // /Dh, %Dh
      out[(((size_t)bb * Hc + h) * Tc + t) * Dh + d] = v;
    }
  }
}

// ---------------------------------------------------------------------------
// scores[bh, t, s] = (q[bh,t,:] . kfull[bh,s,:]) / sqrt(Dh)
// kfull = concat(cache_keys, knew). One block (8 waves) per (b,h); each wave
// walks s-chunks of 16; K=64 -> two 16x16x32 WMMAs per chunk.
// ---------------------------------------------------------------------------
__global__ __launch_bounds__(256)
void attn_scores(const float* __restrict__ q, const float* __restrict__ ck,
                 const float* __restrict__ knew, float* __restrict__ scores)
{
  const int bh    = blockIdx.x;          // 0 .. B*H-1
  const int wave  = threadIdx.x >> 5;    // 0..7
  const int lane  = threadIdx.x & 31;
  const int lrow  = lane & 15;
  const int khalf = lane >> 4;

  const bool qvalid = lrow < Tc;
  const float* __restrict__ qrow = q + ((size_t)bh * Tc + lrow) * Dh;
  const float4 f4z = make_float4(0.f, 0.f, 0.f, 0.f);

  v16bf aq[2];
#pragma unroll
  for (int kb = 0; kb < 2; ++kb) {
    const int base = kb * 32 + khalf * 8;
    put4(aq[kb], 0,  qvalid ? ld4(qrow + base)      : f4z);
    put4(aq[kb], 4,  qvalid ? ld4(qrow + base + 4)  : f4z);
    put4(aq[kb], 8,  qvalid ? ld4(qrow + base + 16) : f4z);
    put4(aq[kb], 12, qvalid ? ld4(qrow + base + 20) : f4z);
  }

  for (int sc = wave; sc < Sc / 16; sc += 8) {
    const int s = sc * 16 + lrow;   // this lane's key index (B-fragment column)
    const float* __restrict__ krow =
        (s < CT) ? ck   + ((size_t)bh * CT + s) * Dh
                 : knew + ((size_t)bh * Tc + (s - CT)) * Dh;
    v8f acc = {};
#pragma unroll
    for (int kb = 0; kb < 2; ++kb) {
      const int base = kb * 32 + khalf * 8;
      v16bf bk;
      put4(bk, 0,  ld4(krow + base));
      put4(bk, 4,  ld4(krow + base + 4));
      put4(bk, 8,  ld4(krow + base + 16));
      put4(bk, 12, ld4(krow + base + 20));
      acc = __builtin_amdgcn_wmma_f32_16x16x32_bf16(false, aq[kb], false, bk,
                                                    (short)0, acc, false, false);
    }
    if (khalf == 0) {   // lanes 0-15 hold rows m=0..7 (the real T rows)
#pragma unroll
      for (int r = 0; r < 8; ++r)
        scores[((size_t)bh * Tc + r) * Sc + s] = acc[r] * 0.125f; // 1/sqrt(64)
    }
  }
}

// ---------------------------------------------------------------------------
// In-place softmax over s for each (b,h,t) row. One wave per row, float4 wide.
// ---------------------------------------------------------------------------
__global__ __launch_bounds__(32)
void softmax_rows(float* __restrict__ scores)
{
  float* __restrict__ p = scores + (size_t)blockIdx.x * Sc;
  const int lane = threadIdx.x;

  float m = -3.0e38f;
  for (int i = lane * 4; i < Sc; i += 128) {
    const float4 f = ld4(p + i);
    m = fmaxf(m, fmaxf(fmaxf(f.x, f.y), fmaxf(f.z, f.w)));
  }
#pragma unroll
  for (int off = 16; off > 0; off >>= 1) m = fmaxf(m, __shfl_xor(m, off, 32));

  float sum = 0.0f;
  for (int i = lane * 4; i < Sc; i += 128) {
    float4 f = ld4(p + i);
    f.x = __expf(f.x - m); f.y = __expf(f.y - m);
    f.z = __expf(f.z - m); f.w = __expf(f.w - m);
    st4(p + i, f);
    sum += (f.x + f.y) + (f.z + f.w);
  }
#pragma unroll
  for (int off = 16; off > 0; off >>= 1) sum += __shfl_xor(sum, off, 32);

  const float inv = 1.0f / sum;
  for (int i = lane * 4; i < Sc; i += 128) {
    float4 f = ld4(p + i);
    f.x *= inv; f.y *= inv; f.z *= inv; f.w *= inv;
    st4(p + i, f);
  }
}

// ---------------------------------------------------------------------------
// ctx[b, t, h*Dh + d] = sum_s P[bh,t,s] * vfull[bh,s,d]
// One wave per (b,h). V chunks (32 x 64 fp32 = 8KB) are staged into LDS with
// async global->LDS b128 copies, double-buffered on ASYNCcnt; B-fragments are
// then built from LDS (ds_load), so the global side stays fully coalesced.
// ---------------------------------------------------------------------------
__global__ __launch_bounds__(32)
void attn_pv(const float* __restrict__ scores, const float* __restrict__ cv,
             const float* __restrict__ vnew, float* __restrict__ ctx)
{
  __shared__ float vtile[2][32 * Dh];   // 2 x 8KB double buffer

  const int bh = blockIdx.x;
  const int b  = bh / Hc, h = bh % Hc;
  const int lane  = threadIdx.x & 31;
  const int lrow  = lane & 15;
  const int khalf = lane >> 4;

  const bool pvalid = lrow < Tc;
  const float* __restrict__ prow = scores + ((size_t)bh * Tc + lrow) * Sc;
  const float4 f4z = make_float4(0.f, 0.f, 0.f, 0.f);

  // Issue 16 async b128 copies (one 32x64 chunk) into buffer `buf`.
  auto issue_chunk = [&](int buf, int s0) {
#pragma unroll
    for (int it = 0; it < 16; ++it) {
      const int idx = it * 32 + lane;     // 0..511 b128 slots
      const int row = idx >> 4;           // 0..31
      const int c4  = (idx & 15) * 4;     // float column (0,4,...,60)
      const int s   = s0 + row;
      const float* src = (s < CT)
          ? cv   + ((size_t)bh * CT + s) * Dh + c4
          : vnew + ((size_t)bh * Tc + (s - CT)) * Dh + c4;
      const unsigned dst = (unsigned)(uintptr_t)&vtile[buf][row * Dh + c4];
      async_ld_f4_to_lds(dst, src);
    }
  };

  v8f acc[4] = {v8f{}, v8f{}, v8f{}, v8f{}};

  issue_chunk(0, 0);
  for (int s0 = 0; s0 < Sc; s0 += 32) {
    const int cur = (s0 >> 5) & 1;
    const bool more = (s0 + 32) < Sc;
    if (more) issue_chunk(1 - cur, s0 + 32);
    if (more) wait_async_le16();   // chunk `cur` (16 in-order ops) has landed
    else      wait_async_0();

    // A fragment from probabilities P (rows t, K = s) -- global, coalesced
    v16bf ap;
    {
      const int base = s0 + khalf * 8;
      put4(ap, 0,  pvalid ? ld4(prow + base)      : f4z);
      put4(ap, 4,  pvalid ? ld4(prow + base + 4)  : f4z);
      put4(ap, 8,  pvalid ? ld4(prow + base + 16) : f4z);
      put4(ap, 12, pvalid ? ld4(prow + base + 20) : f4z);
    }
    // B fragments from staged V (K = s rows, N = d columns) -- LDS reads
    const float* __restrict__ vt = &vtile[cur][0];
#pragma unroll
    for (int nt = 0; nt < 4; ++nt) {
      const int d = nt * 16 + lrow;
      v16bf bvv;
#pragma unroll
      for (int i = 0; i < 16; ++i)
        bvv[i] = (__bf16)vt[(khalf * 8 + (i < 8 ? i : i + 8)) * Dh + d];
      acc[nt] = __builtin_amdgcn_wmma_f32_16x16x32_bf16(false, ap, false, bvv,
                                                        (short)0, acc[nt],
                                                        false, false);
    }
  }

  if (khalf == 0) {   // rows t = 0..7
#pragma unroll
    for (int nt = 0; nt < 4; ++nt) {
#pragma unroll
      for (int r = 0; r < 8; ++r) {
        const int d = nt * 16 + lrow;
        ctx[((size_t)b * Tc + r) * Dm + h * Dh + d] = acc[nt][r];
      }
    }
  }
}

// ---------------------------------------------------------------------------
// Host side: seven launches on `stream`.
// Workspace (floats): q | knew | vnew | ctx | scores  (~42 MB total)
// ---------------------------------------------------------------------------
extern "C" void kernel_launch(void* const* d_in, const int* in_sizes, int n_in,
                              void* d_out, int out_size, void* d_ws, size_t ws_size,
                              hipStream_t stream)
{
  const float* x  = (const float*)d_in[0];
  const float* ck = (const float*)d_in[1];
  const float* cv = (const float*)d_in[2];
  const float* Wq = (const float*)d_in[3];
  const float* bq = (const float*)d_in[4];
  const float* Wk = (const float*)d_in[5];
  const float* bk = (const float*)d_in[6];
  const float* Wv = (const float*)d_in[7];
  const float* bv = (const float*)d_in[8];
  const float* Wo = (const float*)d_in[9];
  const float* bo = (const float*)d_in[10];
  float* out = (float*)d_out;

  const size_t qkvElems = (size_t)Bc * Hc * Tc * Dh;  // 524288
  float* ws     = (float*)d_ws;
  float* q      = ws;
  float* knew   = q    + qkvElems;
  float* vnew   = knew + qkvElems;
  float* ctx    = vnew + qkvElems;
  float* scores = ctx  + qkvElems;                    // B*H*T*S = 8388608

  const dim3 gGemm(Dm / 16, Mrows / 16);  // 128 x 16 tiles

  gemm_xwT_bf16<1><<<gGemm, 32, 0, stream>>>(x,  Wq, bq, q);
  gemm_xwT_bf16<1><<<gGemm, 32, 0, stream>>>(x,  Wk, bk, knew);
  gemm_xwT_bf16<1><<<gGemm, 32, 0, stream>>>(x,  Wv, bv, vnew);

  attn_scores <<<Bc * Hc,      256, 0, stream>>>(q, ck, knew, scores);
  softmax_rows<<<Bc * Hc * Tc,  32, 0, stream>>>(scores);
  attn_pv     <<<Bc * Hc,       32, 0, stream>>>(scores, cv, vnew, ctx);

  gemm_xwT_bf16<0><<<gGemm, 32, 0, stream>>>(ctx, Wo, bo, out);
}
